// Block_59090160058987
// MI455X (gfx1250) — compile-verified, matrix operand order checked by reading, not run on hardware
//
#include <hip/hip_runtime.h>
#include <cmath>

#define S_LEN 2048
#define DMODEL 768
#define NHD 24
#define QNOPE 32
#define QROPE 16
#define QHD 48
#define VHD 32
#define QLR 384
#define KVLR 128
#define PKH 4
#define KDIM 128
#define NEXP 16384
#define KNN 8
#define ACT_HID 24
#define SHID 3072
#define EPSV 1e-5f

typedef __attribute__((ext_vector_type(16))) __bf16 bf16x16;
typedef __attribute__((ext_vector_type(8)))  float  floatx8;
typedef __attribute__((ext_vector_type(4)))  float  f32x4;
typedef __attribute__((ext_vector_type(4)))  unsigned int u32x4;
typedef int v4i __attribute__((vector_size(4 * sizeof(int))));  // matches builtin proto

union FragAB { bf16x16 v; f32x4 q[2]; __bf16 h[16]; };
union FragAcc { floatx8 v; float f[8]; };

// Async memory->LDS copy (gfx1250 GLOBAL_LOAD_ASYNC_TO_LDS_B128, ASYNCcnt),
// with a synchronous fallback if the toolchain lacks the builtins.
#if defined(__gfx1250__) && __has_builtin(__builtin_amdgcn_global_load_async_to_lds_b128) && __has_builtin(__builtin_amdgcn_s_wait_asynccnt)
#define USE_ASYNC 1
#else
#define USE_ASYNC 0
#endif

__device__ __forceinline__ void cp16(const __bf16* g, __bf16* l) {
#if USE_ASYNC
  __builtin_amdgcn_global_load_async_to_lds_b128(
      (__attribute__((address_space(1))) v4i*)g,
      (__attribute__((address_space(3))) v4i*)l, 0, 0);
#else
  *(u32x4*)l = *(const u32x4*)g;
#endif
}
__device__ __forceinline__ void cp_wait() {
#if USE_ASYNC
  __builtin_amdgcn_s_wait_asynccnt(0);
#endif
}

// ---------------------------------------------------------------------------
// WMMA GEMM: C[M,N] = A[M,K] @ W[N,K]^T, bf16 inputs, f32 accum.
// 256 thr (8 waves), 128x128 tile, K-step 32, double-buffered LDS,
// async global->LDS tile DMA, one barrier per K-step.
// ---------------------------------------------------------------------------
__device__ __forceinline__ void load_tiles(
    const __bf16* __restrict__ A, const __bf16* __restrict__ W,
    __bf16 (*sa)[40], __bf16 (*sb)[40],
    int bm, int bn, int M, int N, int K, int k0, int tid) {
#pragma unroll
  for (int c = 0; c < 2; ++c) {
    int chunk = tid + c * 256;      // 0..511 : 16B segments of a 128x32 tile
    int r = chunk >> 2;
    int seg = (chunk & 3) << 3;     // bf16 offset in row
    int ga = bm + r; if (ga >= M) ga = M - 1;   // clamp: rows only feed
    int gb = bn + r; if (gb >= N) gb = N - 1;   // guarded-out outputs
    cp16(&A[(size_t)ga * K + k0 + seg], &sa[r][seg]);
    cp16(&W[(size_t)gb * K + k0 + seg], &sb[r][seg]);
  }
}

__global__ __launch_bounds__(256) void k_gemm(
    const __bf16* __restrict__ A, const __bf16* __restrict__ W,
    float* __restrict__ C, int M, int N, int K,
    const float* __restrict__ bias, const float* __restrict__ bng,
    const float* __restrict__ bnb, const float* __restrict__ res, float scale) {
  __shared__ __align__(16) __bf16 sA[2][128][40];  // 80B row stride: aligned b128
  __shared__ __align__(16) __bf16 sB[2][128][40];
  const int bm = blockIdx.y * 128;
  const int bn = blockIdx.x * 128;
  const int tid = threadIdx.x;
  const int wave = tid >> 5;
  const int lane = tid & 31;
  const int lr = lane & 15;
  const int lh = lane >> 4;   // wave32 half
  const int kb = lh * 8;      // K-base of 16-bit WMMA A/B layout
  const int wm = wave & 3;    // 4 waves over M (32 rows each)
  const int wn = wave >> 2;   // 2 waves over N (64 cols each)

  FragAcc acc[2][4];
#pragma unroll
  for (int mi = 0; mi < 2; ++mi)
#pragma unroll
    for (int ni = 0; ni < 4; ++ni)
#pragma unroll
      for (int i = 0; i < 8; ++i) acc[mi][ni].f[i] = 0.f;

  load_tiles(A, W, sA[0], sB[0], bm, bn, M, N, K, 0, tid);
  int buf = 0;
  for (int k0 = 0; k0 < K; k0 += 32) {
    cp_wait();
    __syncthreads();   // tiles for this step visible; prev step fully consumed
    if (k0 + 32 < K)
      load_tiles(A, W, sA[buf ^ 1], sB[buf ^ 1], bm, bn, M, N, K, k0 + 32, tid);
    FragAB fa[2], fb[4];
#pragma unroll
    for (int mi = 0; mi < 2; ++mi) {
      int rb = wm * 32 + mi * 16 + lr;
      fa[mi].q[0] = *(const f32x4*)&sA[buf][rb][kb];
      fa[mi].q[1] = *(const f32x4*)&sA[buf][rb][16 + kb];
    }
#pragma unroll
    for (int ni = 0; ni < 4; ++ni) {
      int cb = wn * 64 + ni * 16 + lr;
      fb[ni].q[0] = *(const f32x4*)&sB[buf][cb][kb];
      fb[ni].q[1] = *(const f32x4*)&sB[buf][cb][16 + kb];
    }
#pragma unroll
    for (int mi = 0; mi < 2; ++mi)
#pragma unroll
      for (int ni = 0; ni < 4; ++ni)
        acc[mi][ni].v = __builtin_amdgcn_wmma_f32_16x16x32_bf16(
            false, fa[mi].v, false, fb[ni].v, (short)0, acc[mi][ni].v, false, false);
    buf ^= 1;
  }
#pragma unroll
  for (int mi = 0; mi < 2; ++mi)
#pragma unroll
    for (int ni = 0; ni < 4; ++ni)
#pragma unroll
      for (int i = 0; i < 8; ++i) {
        int row = bm + wm * 32 + mi * 16 + i + 8 * lh;
        int col = bn + wn * 64 + ni * 16 + lr;
        if (row < M && col < N) {
          float v = acc[mi][ni].f[i];
          if (bias) v += bias[col];
          if (bng)  v = v * scale * bng[col] + bnb[col];
          if (res)  v += res[(size_t)row * N + col];
          C[(size_t)row * N + col] = v;
        }
      }
}

// ---------------------------------------------------------------------------
// fp32 -> bf16 cast (weights, once per call)
// ---------------------------------------------------------------------------
__global__ void k_cast(const float* __restrict__ in, __bf16* __restrict__ out, int n4) {
  int i = blockIdx.x * blockDim.x + threadIdx.x;
  if (i < n4) {
    f32x4 v = *(const f32x4*)&in[i * 4];
    out[i * 4 + 0] = (__bf16)v.x; out[i * 4 + 1] = (__bf16)v.y;
    out[i * 4 + 2] = (__bf16)v.z; out[i * 4 + 3] = (__bf16)v.w;
  }
}

// ---------------------------------------------------------------------------
// RMSNorm: one block per row; optional fp32 and bf16 outputs
// ---------------------------------------------------------------------------
__global__ __launch_bounds__(256) void k_rmsnorm(
    const float* __restrict__ in, const float* __restrict__ w,
    float* __restrict__ out, __bf16* __restrict__ outb,
    int cols, int istride, int ostride) {
  const int row = blockIdx.x;
  const float* ip = in + (size_t)row * istride;
  float ss = 0.f;
  for (int c = threadIdx.x; c < cols; c += 256) { float v = ip[c]; ss += v * v; }
#pragma unroll
  for (int d = 1; d < 32; d <<= 1) ss += __shfl_xor(ss, d, 32);
  __shared__ float red[8];
  if ((threadIdx.x & 31) == 0) red[threadIdx.x >> 5] = ss;
  __syncthreads();
  if (threadIdx.x == 0) {
    float t = 0.f;
    for (int i = 0; i < 8; ++i) t += red[i];
    red[0] = rsqrtf(t / (float)cols + EPSV);
  }
  __syncthreads();
  float r = red[0];
  for (int c = threadIdx.x; c < cols; c += 256) {
    float v = ip[c] * r * w[c];
    if (out)  out[(size_t)row * ostride + c] = v;
    if (outb) outb[(size_t)row * ostride + c] = (__bf16)v;
  }
}

// ---------------------------------------------------------------------------
// Build per-head q_states/k_states/v (RoPE + softmax scale folded into q)
// ---------------------------------------------------------------------------
__global__ void k_prep(const float* __restrict__ q, const float* __restrict__ kv,
                       const float* __restrict__ ckv, const float* __restrict__ fc,
                       const float* __restrict__ fs, float* __restrict__ qs,
                       float* __restrict__ ks, float* __restrict__ vs) {
  int idx = blockIdx.x * blockDim.x + threadIdx.x;
  if (idx >= S_LEN * NHD) return;
  int s = idx / NHD, hd = idx % NHD;
  const float qscale = 0.14433756729740643f;  // 1/sqrt(48)
  const float* qr = q + (size_t)s * (NHD * QHD) + hd * QHD;
  const float* kr = kv + (size_t)s * (NHD * (QNOPE + VHD)) + hd * (QNOPE + VHD);
  float* qo = qs + ((size_t)hd * S_LEN + s) * QHD;
  float* ko = ks + ((size_t)hd * S_LEN + s) * QHD;
  float* vo = vs + ((size_t)hd * S_LEN + s) * VHD;
  for (int d = 0; d < QNOPE; ++d) { qo[d] = qr[d] * qscale; ko[d] = kr[d]; }
  for (int d = 0; d < VHD; ++d) vo[d] = kr[QNOPE + d];
  const float* pe = ckv + (size_t)s * (KVLR + QROPE) + KVLR;
  for (int j = 0; j < QROPE / 2; ++j) {
    float c = fc[s * (QROPE / 2) + j], sn = fs[s * (QROPE / 2) + j];
    float r0 = qr[QNOPE + 2 * j], i0 = qr[QNOPE + 2 * j + 1];
    qo[QNOPE + 2 * j]     = (r0 * c - i0 * sn) * qscale;
    qo[QNOPE + 2 * j + 1] = (r0 * sn + i0 * c) * qscale;
    float r1 = pe[2 * j], i1 = pe[2 * j + 1];
    ko[QNOPE + 2 * j]     = r1 * c - i1 * sn;
    ko[QNOPE + 2 * j + 1] = r1 * sn + i1 * c;
  }
}

// ---------------------------------------------------------------------------
// Flash attention: 1 wave per (head, 16-query tile). QK^T and PV via WMMA.
// Writes bf16 ctx directly (only consumed by the o-proj GEMM).
// ---------------------------------------------------------------------------
__global__ __launch_bounds__(32) void k_attn(
    const float* __restrict__ qs, const float* __restrict__ ks,
    const float* __restrict__ vs, __bf16* __restrict__ ctx) {
  const int qt = blockIdx.x;
  const int hd = blockIdx.y;
  const int lane = threadIdx.x & 31;
  const int lr = lane & 15;
  const int lh = lane >> 4;
  const int kb = lh * 8;
  __shared__ __align__(16) __bf16 pst[16][32];

  const float* qrow = qs + ((size_t)hd * S_LEN + qt * 16 + lr) * QHD;
  FragAB qa0, qa1;
#pragma unroll
  for (int v = 0; v < 8; ++v) {
    int kk = (v < 4 ? 0 : 16) + kb + 2 * (v & 3);
    qa0.h[2 * v]     = (__bf16)qrow[kk];
    qa0.h[2 * v + 1] = (__bf16)qrow[kk + 1];
    if (v < 4) {
      qa1.h[2 * v]     = (__bf16)qrow[32 + kk];
      qa1.h[2 * v + 1] = (__bf16)qrow[32 + kk + 1];
    } else { qa1.h[2 * v] = (__bf16)0.f; qa1.h[2 * v + 1] = (__bf16)0.f; }
  }

  FragAcc o0, o1;
  float m[8], lsum[8];
#pragma unroll
  for (int i = 0; i < 8; ++i) { o0.f[i] = 0.f; o1.f[i] = 0.f; m[i] = -3e38f; lsum[i] = 0.f; }

  const int qmax = qt * 16 + 15;
  const int nkb = (qmax >> 5) + 1;
  for (int j = 0; j < nkb; ++j) {
    FragAcc s0, s1;
#pragma unroll
    for (int i = 0; i < 8; ++i) { s0.f[i] = 0.f; s1.f[i] = 0.f; }
#pragma unroll
    for (int g = 0; g < 2; ++g) {
      const float* krow = ks + ((size_t)hd * S_LEN + j * 32 + g * 16 + lr) * QHD;
      FragAB kf0, kf1;
#pragma unroll
      for (int v = 0; v < 8; ++v) {
        int kk = (v < 4 ? 0 : 16) + kb + 2 * (v & 3);
        kf0.h[2 * v]     = (__bf16)krow[kk];
        kf0.h[2 * v + 1] = (__bf16)krow[kk + 1];
        if (v < 4) {
          kf1.h[2 * v]     = (__bf16)krow[32 + kk];
          kf1.h[2 * v + 1] = (__bf16)krow[32 + kk + 1];
        } else { kf1.h[2 * v] = (__bf16)0.f; kf1.h[2 * v + 1] = (__bf16)0.f; }
      }
      FragAcc* sp = g ? &s1 : &s0;
      sp->v = __builtin_amdgcn_wmma_f32_16x16x32_bf16(false, qa0.v, false, kf0.v, (short)0, sp->v, false, false);
      sp->v = __builtin_amdgcn_wmma_f32_16x16x32_bf16(false, qa1.v, false, kf1.v, (short)0, sp->v, false, false);
    }
    float mnew[8];
#pragma unroll
    for (int i = 0; i < 8; ++i) {
      int row = qt * 16 + i + 8 * lh;
      int key0 = j * 32 + lr;
      if (key0 > row)      s0.f[i] = -1e9f;
      if (key0 + 16 > row) s1.f[i] = -1e9f;
      mnew[i] = fmaxf(m[i], fmaxf(s0.f[i], s1.f[i]));
    }
#pragma unroll
    for (int d = 1; d < 16; d <<= 1)
#pragma unroll
      for (int i = 0; i < 8; ++i) mnew[i] = fmaxf(mnew[i], __shfl_xor(mnew[i], d, 32));
    float p0[8], p1[8];
#pragma unroll
    for (int i = 0; i < 8; ++i) {
      float alpha = __expf(m[i] - mnew[i]);
      m[i] = mnew[i];
      lsum[i] *= alpha;
      o0.f[i] *= alpha; o1.f[i] *= alpha;
      p0[i] = __expf(s0.f[i] - mnew[i]);
      p1[i] = __expf(s1.f[i] - mnew[i]);
      lsum[i] += p0[i] + p1[i];
    }
#pragma unroll
    for (int i = 0; i < 8; ++i) {
      pst[i + 8 * lh][lr]      = (__bf16)p0[i];
      pst[i + 8 * lh][16 + lr] = (__bf16)p1[i];
    }
    __syncthreads();
    FragAB pa;
    pa.q[0] = *(const f32x4*)&pst[lr][kb];
    pa.q[1] = *(const f32x4*)&pst[lr][16 + kb];
    FragAB vf0, vf1;
#pragma unroll
    for (int v = 0; v < 8; ++v) {
      int kk = (v < 4 ? 0 : 16) + kb + 2 * (v & 3);
      const float* vr = vs + ((size_t)hd * S_LEN + j * 32 + kk) * VHD;
      vf0.h[2 * v]     = (__bf16)vr[lr];
      vf0.h[2 * v + 1] = (__bf16)vr[VHD + lr];
      vf1.h[2 * v]     = (__bf16)vr[16 + lr];
      vf1.h[2 * v + 1] = (__bf16)vr[VHD + 16 + lr];
    }
    o0.v = __builtin_amdgcn_wmma_f32_16x16x32_bf16(false, pa.v, false, vf0.v, (short)0, o0.v, false, false);
    o1.v = __builtin_amdgcn_wmma_f32_16x16x32_bf16(false, pa.v, false, vf1.v, (short)0, o1.v, false, false);
    __syncthreads();
  }
#pragma unroll
  for (int d = 1; d < 16; d <<= 1)
#pragma unroll
    for (int i = 0; i < 8; ++i) lsum[i] += __shfl_xor(lsum[i], d, 32);
#pragma unroll
  for (int i = 0; i < 8; ++i) {
    int row = qt * 16 + i + 8 * lh;
    float inv = 1.f / lsum[i];
    ctx[(size_t)row * (NHD * VHD) + hd * VHD + lr]      = (__bf16)(o0.f[i] * inv);
    ctx[(size_t)row * (NHD * VHD) + hd * VHD + 16 + lr] = (__bf16)(o1.f[i] * inv);
  }
}

// ---------------------------------------------------------------------------
// Fused PK scoring + top-8 + softmax (keys stream through 192MB L2)
// ---------------------------------------------------------------------------
__device__ inline void top8_ins(float* s, int* id, float sc, int ei) {
  if (sc <= s[7]) return;
#pragma unroll
  for (int t = 7; t >= 1; --t) {
    if (sc > s[t - 1])    { s[t] = s[t - 1]; id[t] = id[t - 1]; }
    else if (sc > s[t])   { s[t] = sc;       id[t] = ei; }
  }
  if (sc > s[0]) { s[0] = sc; id[0] = ei; }
}

__global__ __launch_bounds__(256) void k_topk(
    const float* __restrict__ qf, const float* __restrict__ keys,
    float* __restrict__ topw, int* __restrict__ topi) {
  const int s = blockIdx.x;
  const int hd = blockIdx.y;
  const int tid = threadIdx.x;
  __shared__ __align__(16) float qrow[KDIM];
  __shared__ float sb[2048];
  __shared__ int ib[2048];
  if (tid < KDIM) qrow[tid] = qf[(size_t)s * (PKH * KDIM) + hd * KDIM + tid];
  __syncthreads();
  float bs[8]; int bi[8];
#pragma unroll
  for (int t = 0; t < 8; ++t) { bs[t] = -3e38f; bi[t] = 0; }
  const f32x4* qv = (const f32x4*)qrow;
  for (int e = tid; e < NEXP; e += 256) {
    const f32x4* kp = (const f32x4*)(keys + ((size_t)hd * NEXP + e) * KDIM);
    float acc = 0.f;
#pragma unroll 8
    for (int t = 0; t < KDIM / 4; ++t) {
      f32x4 a = qv[t], b = kp[t];
      acc += a.x * b.x + a.y * b.y + a.z * b.z + a.w * b.w;
    }
    top8_ins(bs, bi, acc, e);
  }
#pragma unroll
  for (int t = 0; t < 8; ++t) { sb[tid * 8 + t] = bs[t]; ib[tid * 8 + t] = bi[t]; }
  __syncthreads();
  if (tid < 32) {
    float ms[8]; int mi[8];
#pragma unroll
    for (int t = 0; t < 8; ++t) { ms[t] = -3e38f; mi[t] = 0; }
    for (int j = tid * 64; j < tid * 64 + 64; ++j) top8_ins(ms, mi, sb[j], ib[j]);
#pragma unroll
    for (int d = 16; d >= 1; d >>= 1) {
      float os[8]; int oi[8];
#pragma unroll
      for (int t = 0; t < 8; ++t) { os[t] = __shfl_xor(ms[t], d, 32); oi[t] = __shfl_xor(mi[t], d, 32); }
#pragma unroll
      for (int t = 0; t < 8; ++t) top8_ins(ms, mi, os[t], oi[t]);
    }
    if (tid == 0) {
      float mx = ms[0], sum = 0.f, e8[8];
#pragma unroll
      for (int t = 0; t < 8; ++t) { e8[t] = __expf(ms[t] - mx); sum += e8[t]; }
      float inv = 1.f / sum;
#pragma unroll
      for (int t = 0; t < 8; ++t) {
        topw[((size_t)s * PKH + hd) * KNN + t] = e8[t] * inv;
        topi[((size_t)s * PKH + hd) * KNN + t] = mi[t];
      }
    }
  }
}

// ---------------------------------------------------------------------------
// Per-token MoE expert apply
// ---------------------------------------------------------------------------
__global__ __launch_bounds__(256) void k_expert(
    const float* __restrict__ y, const float* __restrict__ wdn,
    const float* __restrict__ wup, const float* __restrict__ topw,
    const int* __restrict__ topi, const float* __restrict__ aw1,
    const float* __restrict__ aw2, const float* __restrict__ aw3,
    float* __restrict__ moe) {
  const int s = blockIdx.x;
  const int tid = threadIdx.x;
  __shared__ __align__(16) float yrow[DMODEL];
  __shared__ float zls[32];
  __shared__ float hid[PKH * ACT_HID];
  __shared__ float zout[32];
  __shared__ int ix[32];
  for (int d = tid; d < DMODEL; d += 256) yrow[d] = y[(size_t)s * DMODEL + d];
  if (tid < 32) ix[tid] = topi[(size_t)s * 32 + tid];
  __syncthreads();
  {
    int p = tid >> 3, q = tid & 7;
    const float* wr = wdn + (size_t)ix[p] * DMODEL;
    float acc = 0.f;
    for (int d = q; d < DMODEL; d += 8) acc += yrow[d] * wr[d];
    acc += __shfl_xor(acc, 1, 32);
    acc += __shfl_xor(acc, 2, 32);
    acc += __shfl_xor(acc, 4, 32);
    if (q == 0) zls[p] = acc;
  }
  __syncthreads();
  if (tid < PKH * ACT_HID) {
    int h4 = tid / ACT_HID, j = tid % ACT_HID;
    float a = 0.f, b = 0.f;
#pragma unroll
    for (int k = 0; k < KNN; ++k) {
      float z = zls[h4 * KNN + k];
      a += z * aw1[j * KNN + k];
      b += z * aw3[j * KNN + k];
    }
    hid[h4 * ACT_HID + j] = (a / (1.f + __expf(-a))) * b;
  }
  __syncthreads();
  if (tid < 32) {
    int h4 = tid >> 3, k = tid & 7;
    float o = 0.f;
#pragma unroll
    for (int j = 0; j < ACT_HID; ++j) o += hid[h4 * ACT_HID + j] * aw2[k * ACT_HID + j];
    zout[tid] = o * topw[(size_t)s * 32 + tid];
  }
  __syncthreads();
  for (int d = tid; d < DMODEL; d += 256) {
    float acc = 0.f;
    for (int p = 0; p < 32; ++p) acc += zout[p] * wup[(size_t)ix[p] * DMODEL + d];
    moe[(size_t)s * DMODEL + d] = acc;
  }
}

__global__ void k_silu_mul(const float* __restrict__ a, const float* __restrict__ b,
                           __bf16* __restrict__ o, int n) {
  int i = blockIdx.x * blockDim.x + threadIdx.x;
  if (i < n) { float x = a[i]; o[i] = (__bf16)((x / (1.f + __expf(-x))) * b[i]); }
}

__global__ void k_final(const float* __restrict__ h, const float* __restrict__ moe,
                        const float* __restrict__ sh, float* __restrict__ out, int n) {
  int i = blockIdx.x * blockDim.x + threadIdx.x;
  if (i < n) out[i] = h[i] + moe[i] + sh[i];
}

// ---------------------------------------------------------------------------
extern "C" void kernel_launch(void* const* d_in, const int* in_sizes, int n_in,
                              void* d_out, int out_size, void* d_ws, size_t ws_size,
                              hipStream_t stream) {
  (void)in_sizes; (void)n_in; (void)out_size; (void)ws_size;
  const float* x         = (const float*)d_in[0];
  const float* norm1_w   = (const float*)d_in[1];
  const float* q_a_w     = (const float*)d_in[2];
  const float* q_a_ln_w  = (const float*)d_in[3];
  const float* q_b_w     = (const float*)d_in[4];
  const float* kv_a_w    = (const float*)d_in[5];
  const float* kv_a_ln_w = (const float*)d_in[6];
  const float* kv_b_w    = (const float*)d_in[7];
  const float* o_w       = (const float*)d_in[8];
  const float* norm2_w   = (const float*)d_in[9];
  const float* qp_w      = (const float*)d_in[10];
  const float* qp_b      = (const float*)d_in[11];
  const float* bn_g      = (const float*)d_in[12];
  const float* bn_b      = (const float*)d_in[13];
  const float* keys      = (const float*)d_in[14];
  const float* aw1       = (const float*)d_in[15];
  const float* aw2       = (const float*)d_in[16];
  const float* aw3       = (const float*)d_in[17];
  const float* w_down    = (const float*)d_in[18];
  const float* w_up      = (const float*)d_in[19];
  const float* sw1       = (const float*)d_in[20];
  const float* sw2       = (const float*)d_in[21];
  const float* sw3       = (const float*)d_in[22];
  const float* fc        = (const float*)d_in[24];
  const float* fs        = (const float*)d_in[25];
  float* out = (float*)d_out;

  char* cb = (char*)d_ws;
  size_t off = 0;
  auto allocB = [&](size_t bytes) {
    void* p = cb + off;
    off += (bytes + 255) & ~(size_t)255;
    return p;
  };
  auto allocF = [&](size_t n) { return (float*)allocB(n * 4); };
  auto allocH = [&](size_t n) { return (__bf16*)allocB(n * 2); };

  // fp32 intermediates
  float* qa    = allocF((size_t)S_LEN * QLR);
  float* qbuf  = allocF((size_t)S_LEN * NHD * QHD);
  float* ckv   = allocF((size_t)S_LEN * (KVLR + QROPE));
  float* kvb   = allocF((size_t)S_LEN * NHD * (QNOPE + VHD));
  float* qst   = allocF((size_t)NHD * S_LEN * QHD);
  float* kst   = allocF((size_t)NHD * S_LEN * QHD);
  float* vst   = allocF((size_t)NHD * S_LEN * VHD);
  float* hbuf  = allocF((size_t)S_LEN * DMODEL);
  float* ybuf  = allocF((size_t)S_LEN * DMODEL);
  float* qf    = allocF((size_t)S_LEN * PKH * KDIM);
  float* topw  = allocF((size_t)S_LEN * 32);
  int*   topi  = (int*)allocB((size_t)S_LEN * 32 * 4);
  float* g1    = allocF((size_t)S_LEN * SHID);
  float* g3    = allocF((size_t)S_LEN * SHID);
  float* shd   = allocF((size_t)S_LEN * DMODEL);
  float* moe   = allocF((size_t)S_LEN * DMODEL);
  // bf16 activations
  __bf16* hinb = allocH((size_t)S_LEN * DMODEL);
  __bf16* qanb = allocH((size_t)S_LEN * QLR);
  __bf16* cknb = allocH((size_t)S_LEN * KVLR);
  __bf16* ctxb = allocH((size_t)S_LEN * DMODEL);
  __bf16* ybb  = allocH((size_t)S_LEN * DMODEL);
  __bf16* g1b  = allocH((size_t)S_LEN * SHID);
  // bf16 weights
  __bf16* wqa  = allocH((size_t)QLR * DMODEL);
  __bf16* wqb  = allocH((size_t)NHD * QHD * QLR);
  __bf16* wkva = allocH((size_t)(KVLR + QROPE) * DMODEL);
  __bf16* wkvb = allocH((size_t)NHD * (QNOPE + VHD) * KVLR);
  __bf16* wo   = allocH((size_t)DMODEL * DMODEL);
  __bf16* wqp  = allocH((size_t)PKH * KDIM * DMODEL);
  __bf16* ws1  = allocH((size_t)SHID * DMODEL);
  __bf16* ws2  = allocH((size_t)DMODEL * SHID);
  __bf16* ws3  = allocH((size_t)SHID * DMODEL);

  dim3 b256(256);
  auto cast = [&](const float* in, __bf16* o, size_t n) {
    k_cast<<<(unsigned)((n / 4 + 255) / 256), b256, 0, stream>>>(in, o, (int)(n / 4));
  };
  auto gemm = [&](const __bf16* A, const __bf16* W, float* C, int M, int N, int K,
                  const float* bias, const float* g, const float* bb2,
                  const float* res, float scale) {
    dim3 grid((N + 127) / 128, (M + 127) / 128);
    k_gemm<<<grid, b256, 0, stream>>>(A, W, C, M, N, K, bias, g, bb2, res, scale);
  };

  // 0) weights -> bf16 (once per call; deterministic)
  cast(q_a_w, wqa, (size_t)QLR * DMODEL);
  cast(q_b_w, wqb, (size_t)NHD * QHD * QLR);
  cast(kv_a_w, wkva, (size_t)(KVLR + QROPE) * DMODEL);
  cast(kv_b_w, wkvb, (size_t)NHD * (QNOPE + VHD) * KVLR);
  cast(o_w, wo, (size_t)DMODEL * DMODEL);
  cast(qp_w, wqp, (size_t)PKH * KDIM * DMODEL);
  cast(sw1, ws1, (size_t)SHID * DMODEL);
  cast(sw2, ws2, (size_t)DMODEL * SHID);
  cast(sw3, ws3, (size_t)SHID * DMODEL);

  // 1) pre-norm + MLA projections
  k_rmsnorm<<<S_LEN, 256, 0, stream>>>(x, norm1_w, nullptr, hinb, DMODEL, DMODEL, DMODEL);
  gemm(hinb, wqa, qa, S_LEN, QLR, DMODEL, nullptr, nullptr, nullptr, nullptr, 1.f);
  k_rmsnorm<<<S_LEN, 256, 0, stream>>>(qa, q_a_ln_w, nullptr, qanb, QLR, QLR, QLR);
  gemm(qanb, wqb, qbuf, S_LEN, NHD * QHD, QLR, nullptr, nullptr, nullptr, nullptr, 1.f);
  gemm(hinb, wkva, ckv, S_LEN, KVLR + QROPE, DMODEL, nullptr, nullptr, nullptr, nullptr, 1.f);
  k_rmsnorm<<<S_LEN, 256, 0, stream>>>(ckv, kv_a_ln_w, nullptr, cknb, KVLR, KVLR + QROPE, KVLR);
  gemm(cknb, wkvb, kvb, S_LEN, NHD * (QNOPE + VHD), KVLR, nullptr, nullptr, nullptr, nullptr, 1.f);

  // 2) RoPE + head layout, flash attention, output projection (+residual)
  k_prep<<<(S_LEN * NHD + 255) / 256, 256, 0, stream>>>(qbuf, kvb, ckv, fc, fs, qst, kst, vst);
  k_attn<<<dim3(S_LEN / 16, NHD), dim3(32), 0, stream>>>(qst, kst, vst, ctxb);
  gemm(ctxb, wo, hbuf, S_LEN, DMODEL, DMODEL, nullptr, nullptr, nullptr, x, 1.f);

  // 3) post-norm, PK query projection (+bias +BN affine), fused top-k
  k_rmsnorm<<<S_LEN, 256, 0, stream>>>(hbuf, norm2_w, ybuf, ybb, DMODEL, DMODEL, DMODEL);
  float bnscale = 1.0f / sqrtf(1.0f + 1e-5f);
  gemm(ybb, wqp, qf, S_LEN, PKH * KDIM, DMODEL, qp_b, bn_g, bn_b, nullptr, bnscale);
  k_topk<<<dim3(S_LEN, PKH), b256, 0, stream>>>(qf, keys, topw, topi);

  // 4) shared SwiGLU (WMMA GEMMs) + MoE expert apply
  gemm(ybb, ws1, g1, S_LEN, SHID, DMODEL, nullptr, nullptr, nullptr, nullptr, 1.f);
  gemm(ybb, ws3, g3, S_LEN, SHID, DMODEL, nullptr, nullptr, nullptr, nullptr, 1.f);
  k_silu_mul<<<((S_LEN * SHID) + 255) / 256, 256, 0, stream>>>(g1, g3, g1b, S_LEN * SHID);
  gemm(g1b, ws2, shd, S_LEN, DMODEL, SHID, nullptr, nullptr, nullptr, nullptr, 1.f);
  k_expert<<<S_LEN, 256, 0, stream>>>(ybuf, w_down, w_up, topw, topi, aw1, aw2, aw3, moe);

  // 5) out = h + moe + shared
  k_final<<<((S_LEN * DMODEL) + 255) / 256, 256, 0, stream>>>(hbuf, moe, shd, out, S_LEN * DMODEL);
}
